// AttentionAggregator_71871982731886
// MI455X (gfx1250) — compile-verified, matrix operand order checked by reading, not run on hardware
//
#include <hip/hip_runtime.h>
#include <math.h>

#define SLOPE 0.1f
#define IN_DIM 256
#define OUT_DIM 128

typedef __bf16 v16bf __attribute__((ext_vector_type(16)));
typedef float  v8f   __attribute__((ext_vector_type(8)));

// ---------------------------------------------------------------------------
// emb = X @ W^T + b  via v_wmma_f32_16x16x32_bf16, bf16x3 split for accuracy.
// Block: 256 threads = 8 waves. Each wave -> one 16x16 tile; 8 waves cover all
// 128 output columns of a 16-row stripe. K loop: 256 in steps of 32.
// A fragment (16-bit A 16x32): lane<16 -> row=lane, K = k0+[0..7] and k0+16+[0..7]
//                              lane>=16 -> same row set, K bands shifted by 8.
// B[k][n] = W[n][k]  (W row-major [128][256]) -> same per-lane contiguous loads.
// ---------------------------------------------------------------------------
__global__ __launch_bounds__(256)
void gat_gemm_wmma_kernel(const float* __restrict__ X, const float* __restrict__ W,
                          const float* __restrict__ bias, float* __restrict__ emb,
                          int nrows) {
  const int wave = threadIdx.x >> 5;      // 0..7 -> output-column tile
  const int lane = threadIdx.x & 31;
  const int m0   = blockIdx.x << 4;       // 16 rows per block
  const int n0   = wave << 4;             // 16 cols per wave
  const int half = lane >> 4;             // 0: K low band, 1: K+8 band
  const int idx  = lane & 15;

  int mrow = m0 + idx;
  if (mrow >= nrows) mrow = nrows - 1;    // clamp: keep EXEC all-ones for WMMA

  const float* ra = X + (size_t)mrow * IN_DIM + half * 8;
  const float* rb = W + (size_t)(n0 + idx) * IN_DIM + half * 8;

  v8f c = {};
#pragma unroll
  for (int k0 = 0; k0 < IN_DIM; k0 += 32) {
    v16bf ah, al, bh, bl;
#pragma unroll
    for (int e = 0; e < 8; ++e) {
      {
        const float v = ra[k0 + e];
        const __bf16 h = (__bf16)v;
        ah[e] = h; al[e] = (__bf16)(v - (float)h);
      }
      {
        const float v = ra[k0 + 16 + e];
        const __bf16 h = (__bf16)v;
        ah[e + 8] = h; al[e + 8] = (__bf16)(v - (float)h);
      }
      {
        const float v = rb[k0 + e];
        const __bf16 h = (__bf16)v;
        bh[e] = h; bl[e] = (__bf16)(v - (float)h);
      }
      {
        const float v = rb[k0 + 16 + e];
        const __bf16 h = (__bf16)v;
        bh[e + 8] = h; bl[e + 8] = (__bf16)(v - (float)h);
      }
    }
    c = __builtin_amdgcn_wmma_f32_16x16x32_bf16(false, ah, false, bh, (short)0, c, false, false);
    c = __builtin_amdgcn_wmma_f32_16x16x32_bf16(false, ah, false, bl, (short)0, c, false, false);
    c = __builtin_amdgcn_wmma_f32_16x16x32_bf16(false, al, false, bh, (short)0, c, false, false);
  }

  const float bn = bias[n0 + idx];
#pragma unroll
  for (int i = 0; i < 8; ++i) {
    const int m = m0 + half * 8 + i;      // C/D layout: VGPR i -> M=i (lanes<16), M=8+i (lanes>=16)
    if (m < nrows) emb[(size_t)m * OUT_DIM + (n0 + idx)] = c[i] + bn;
  }
}

// ---------------------------------------------------------------------------
// Per-node attention factors: alpha_s[n] = emb[n].a[:128], alpha_t[n] = emb[n].a[128:]
// One wave per node, shuffle reduction (wave32).
// ---------------------------------------------------------------------------
__global__ __launch_bounds__(256)
void gat_alpha_kernel(const float* __restrict__ emb, const float* __restrict__ a,
                      float* __restrict__ alpha_s, float* __restrict__ alpha_t, int n) {
  const int wid  = (int)((blockIdx.x * blockDim.x + threadIdx.x) >> 5);
  const int lane = threadIdx.x & 31;
  if (wid >= n) return;
  const float* row = emb + (size_t)wid * OUT_DIM;
  float s = 0.f, t = 0.f;
#pragma unroll
  for (int i = 0; i < OUT_DIM / 32; ++i) {
    const float v = row[lane + 32 * i];
    s += v * a[lane + 32 * i];
    t += v * a[OUT_DIM + lane + 32 * i];
  }
#pragma unroll
  for (int off = 16; off > 0; off >>= 1) {
    s += __shfl_xor(s, off, 32);
    t += __shfl_xor(t, off, 32);
  }
  if (lane == 0) { alpha_s[wid] = s; alpha_t[wid] = t; }
}

__global__ __launch_bounds__(256)
void gat_zero_kernel(float* __restrict__ p, size_t n) {
  const size_t i = (size_t)blockIdx.x * blockDim.x + threadIdx.x;
  if (i < n) p[i] = 0.f;
}

// ---------------------------------------------------------------------------
// One wave per edge: score = exp(leaky_relu(alpha_s[src] + alpha_t[dst]))
// atomically accumulate score*emb[dst] into num[src] (lives in d_out) and
// score into den[src]. emb (51 MB) resides in the 192 MB L2 -> gathers/atomics
// are L2-resident.
// ---------------------------------------------------------------------------
__global__ __launch_bounds__(256)
void gat_edge_kernel(const float* __restrict__ emb, const int* __restrict__ src,
                     const int* __restrict__ dst, const float* __restrict__ alpha_s,
                     const float* __restrict__ alpha_t, float* __restrict__ num,
                     float* __restrict__ den, int nedges) {
  const int wid  = (int)((blockIdx.x * blockDim.x + threadIdx.x) >> 5);
  const int lane = threadIdx.x & 31;
  if (wid >= nedges) return;
  const int s = src[wid];
  const int t = dst[wid];
  const float logit = alpha_s[s] + alpha_t[t];
  const float lr = (logit > 0.f) ? logit : SLOPE * logit;
  const float sc = expf(lr);
  const float* et = emb + (size_t)t * OUT_DIM;
  float* ns = num + (size_t)s * OUT_DIM;
#pragma unroll
  for (int i = 0; i < OUT_DIM / 32; ++i) {
    const int col = lane + 32 * i;
    __hip_atomic_fetch_add(&ns[col], sc * et[col], __ATOMIC_RELAXED, __HIP_MEMORY_SCOPE_AGENT);
  }
  if (lane == 0)
    __hip_atomic_fetch_add(&den[s], sc, __ATOMIC_RELAXED, __HIP_MEMORY_SCOPE_AGENT);
}

// ---------------------------------------------------------------------------
// Fold in the self-loop edge algebraically and normalize:
// out[n][c] = (num[n][c] + s_n*emb[n][c]) / (den[n] + s_n),
// s_n = exp(leaky_relu(alpha_s[n] + alpha_t[n])).
// ---------------------------------------------------------------------------
__global__ __launch_bounds__(256)
void gat_finalize_kernel(const float* __restrict__ emb, const float* __restrict__ alpha_s,
                         const float* __restrict__ alpha_t, const float* __restrict__ den,
                         float* __restrict__ out, size_t total) {
  const size_t i = (size_t)blockIdx.x * blockDim.x + threadIdx.x;
  if (i >= total) return;
  const size_t node = i >> 7;  // /128
  const float logit = alpha_s[node] + alpha_t[node];
  const float lr = (logit > 0.f) ? logit : SLOPE * logit;
  const float sc = expf(lr);
  out[i] = (out[i] + sc * emb[i]) / (den[node] + sc);
}

extern "C" void kernel_launch(void* const* d_in, const int* in_sizes, int n_in,
                              void* d_out, int out_size, void* d_ws, size_t ws_size,
                              hipStream_t stream) {
  const float* X    = (const float*)d_in[0];   // [N, 256]
  const float* W    = (const float*)d_in[1];   // [128, 256]
  const float* bias = (const float*)d_in[2];   // [128]
  const float* a    = (const float*)d_in[3];   // [256]
  const int*   src  = (const int*)d_in[4];     // [E]
  const int*   dst  = (const int*)d_in[5];     // [E]
  const int N = in_sizes[0] / IN_DIM;
  const int E = in_sizes[4];

  float* emb     = (float*)d_ws;               // N*128
  float* den     = emb + (size_t)N * OUT_DIM;  // N
  float* alpha_s = den + N;                    // N
  float* alpha_t = alpha_s + N;                // N
  float* out     = (float*)d_out;              // N*128 (doubles as num accumulator)

  const size_t total = (size_t)N * OUT_DIM;

  gat_zero_kernel<<<dim3((unsigned)((total + 255) / 256)), dim3(256), 0, stream>>>(out, total);
  gat_zero_kernel<<<dim3((unsigned)((N + 255) / 256)), dim3(256), 0, stream>>>(den, (size_t)N);

  gat_gemm_wmma_kernel<<<dim3((unsigned)((N + 15) / 16)), dim3(256), 0, stream>>>(X, W, bias, emb, N);

  gat_alpha_kernel<<<dim3((unsigned)((N + 7) / 8)), dim3(256), 0, stream>>>(emb, a, alpha_s, alpha_t, N);

  gat_edge_kernel<<<dim3((unsigned)((E + 7) / 8)), dim3(256), 0, stream>>>(
      emb, src, dst, alpha_s, alpha_t, out, den, E);

  gat_finalize_kernel<<<dim3((unsigned)((total + 255) / 256)), dim3(256), 0, stream>>>(
      emb, alpha_s, alpha_t, den, out, total);
}
